// TransformerBlock_71846212927637
// MI455X (gfx1250) — compile-verified, matrix operand order checked by reading, not run on hardware
//
#include <hip/hip_runtime.h>
#include <hip/hip_bf16.h>

#define H  2048
#define S  4096
#define NH 32
#define HD 64
#define FF 8192

typedef __attribute__((ext_vector_type(16))) __bf16 bf16x16;
typedef __attribute__((ext_vector_type(8)))  float  f32x8;

union FragU { uint4 u[2]; bf16x16 v; };

__device__ __forceinline__ unsigned short f2bf(float f) {
    union { float f; unsigned int u; } c; c.f = f;
    unsigned int u = c.u;
    unsigned int r = u + 0x7FFFu + ((u >> 16) & 1u);   // round-to-nearest-even
    return (unsigned short)(r >> 16);
}

// ---------------------------------------------------------------------------
// Weight convert + transpose: W[K][N] fp32 -> WT[N][K] bf16
// ---------------------------------------------------------------------------
__global__ __launch_bounds__(256) void wconvT_kernel(
    const float* __restrict__ W, unsigned short* __restrict__ WT, int K, int N)
{
    __shared__ float tile[32][33];
    int bn = blockIdx.x * 32;           // N tile
    int bk = blockIdx.y * 32;           // K tile
    int tx = threadIdx.x & 31;
    int ty = threadIdx.x >> 5;          // 0..7
    #pragma unroll
    for (int i = ty; i < 32; i += 8)
        tile[i][tx] = W[(size_t)(bk + i) * N + bn + tx];
    __syncthreads();
    #pragma unroll
    for (int i = ty; i < 32; i += 8)
        WT[(size_t)(bn + i) * K + bk + tx] = f2bf(tile[tx][i]);
}

// ---------------------------------------------------------------------------
// LayerNorm over H=2048 cols, one block per row, output bf16
// ---------------------------------------------------------------------------
__global__ __launch_bounds__(256) void lnorm_kernel(
    const float* __restrict__ x, const float* __restrict__ g,
    const float* __restrict__ b, unsigned short* __restrict__ out)
{
    int row = blockIdx.x;
    int t = threadIdx.x, lane = t & 31, wave = t >> 5;
    const float* xr = x + (size_t)row * H;
    float v[8], s = 0.f, s2 = 0.f;
    #pragma unroll
    for (int i = 0; i < 8; ++i) {
        float val = xr[t + i * 256];
        v[i] = val; s += val; s2 += val * val;
    }
    #pragma unroll
    for (int m = 1; m < 32; m <<= 1) { s += __shfl_xor(s, m); s2 += __shfl_xor(s2, m); }
    __shared__ float red[16];
    __shared__ float stats[2];
    if (lane == 0) { red[wave] = s; red[wave + 8] = s2; }
    __syncthreads();
    if (t == 0) {
        float ts = 0.f, ts2 = 0.f;
        #pragma unroll
        for (int i = 0; i < 8; ++i) { ts += red[i]; ts2 += red[i + 8]; }
        float mu = ts * (1.0f / H);
        stats[0] = mu;
        stats[1] = ts2 * (1.0f / H) - mu * mu;
    }
    __syncthreads();
    float mu = stats[0];
    float inv = rsqrtf(stats[1] + 1e-5f);
    #pragma unroll
    for (int i = 0; i < 8; ++i) {
        int c = t + i * 256;
        out[(size_t)row * H + c] = f2bf((v[i] - mu) * inv * g[c] + b[c]);
    }
}

// ---------------------------------------------------------------------------
// bf16 WMMA GEMM: C[M,N] = A[M,K] @ B (B given transposed: BT[N,K])
// block tile 128(M) x 128(N), 8 waves as 4(M) x 2(N), wave tile 32x64
// (2x4 WMMA 16x16 tiles -> 8 v_wmma per K-step per wave)
// mode 0: store bf16 | mode 1: +bias, GELU, bf16 | mode 2: (+bias)+residual, f32
// ---------------------------------------------------------------------------
#define AST 40
#define BST 40

__global__ __launch_bounds__(256) void gemm_bf16_kernel(
    const unsigned short* __restrict__ A,   // M x K
    const unsigned short* __restrict__ BT,  // N x K
    int M, int N, int K, int mode,
    const float* __restrict__ bias, const float* __restrict__ resid,
    unsigned short* __restrict__ Cb, float* __restrict__ Cf)
{
    __shared__ unsigned short At[128 * AST];
    __shared__ unsigned short Bt[128 * BST];
    int m0 = blockIdx.y * 128;
    int n0 = blockIdx.x * 128;
    int t = threadIdx.x, lane = t & 31, wave = t >> 5;
    int wm = wave & 3, wn = wave >> 2;
    int fr = lane & 15, ks = (lane >> 4) * 8;

    f32x8 zero = {0.f,0.f,0.f,0.f,0.f,0.f,0.f,0.f};
    f32x8 acc[2][4];
    #pragma unroll
    for (int mt = 0; mt < 2; ++mt)
        #pragma unroll
        for (int nt = 0; nt < 4; ++nt) acc[mt][nt] = zero;

    int arow = t >> 1, acol = (t & 1) * 16;   // tile loader mapping (128x32 each)

    for (int kk = 0; kk < K; kk += 32) {
        {   // A tile 128x32
            const uint4* src = (const uint4*)(A + (size_t)(m0 + arow) * K + kk + acol);
            uint4* dst = (uint4*)(At + arow * AST + acol);
            dst[0] = src[0];
            dst[1] = src[1];
        }
        {   // B tile 128x32
            const uint4* src = (const uint4*)(BT + (size_t)(n0 + arow) * K + kk + acol);
            uint4* dst = (uint4*)(Bt + arow * BST + acol);
            dst[0] = src[0];
            dst[1] = src[1];
        }
        if (kk + 32 < K) {  // prefetch next K slab toward WGP caches
            __builtin_prefetch(A + (size_t)(m0 + arow) * K + kk + 32 + acol, 0, 3);
            __builtin_prefetch(BT + (size_t)(n0 + arow) * K + kk + 32 + acol, 0, 3);
        }
        __syncthreads();

        FragU a[2], b[4];
        #pragma unroll
        for (int mt = 0; mt < 2; ++mt) {
            const unsigned short* p = At + (wm * 32 + mt * 16 + fr) * AST + ks;
            a[mt].u[0] = *(const uint4*)(p);
            a[mt].u[1] = *(const uint4*)(p + 16);
        }
        #pragma unroll
        for (int nt = 0; nt < 4; ++nt) {
            const unsigned short* p = Bt + (wn * 64 + nt * 16 + fr) * BST + ks;
            b[nt].u[0] = *(const uint4*)(p);
            b[nt].u[1] = *(const uint4*)(p + 16);
        }
        #pragma unroll
        for (int mt = 0; mt < 2; ++mt)
            #pragma unroll
            for (int nt = 0; nt < 4; ++nt)
                acc[mt][nt] = __builtin_amdgcn_wmma_f32_16x16x32_bf16(
                    false, a[mt].v, false, b[nt].v, (short)0, acc[mt][nt], false, false);
        __syncthreads();
    }

    int hi = lane >> 4;
    #pragma unroll
    for (int mt = 0; mt < 2; ++mt)
        #pragma unroll
        for (int nt = 0; nt < 4; ++nt)
            #pragma unroll
            for (int j = 0; j < 8; ++j) {
                int row = m0 + wm * 32 + mt * 16 + hi * 8 + j;
                int col = n0 + wn * 64 + nt * 16 + fr;
                float v = acc[mt][nt][j];
                if (mode == 0) {
                    Cb[(size_t)row * N + col] = f2bf(v);
                } else if (mode == 1) {
                    v += bias[col];
                    v = 0.5f * v * (1.0f + erff(v * 0.70710678118f));
                    Cb[(size_t)row * N + col] = f2bf(v);
                } else {
                    if (bias) v += bias[col];
                    v += resid[(size_t)row * N + col];
                    Cf[(size_t)row * N + col] = v;
                }
            }
}

// ---------------------------------------------------------------------------
// Flash attention, causal + pad mask. One block = 128 query rows of one head,
// 8 waves x 16 rows. Online softmax, all matmuls via v_wmma_f32_16x16x32_bf16.
// q/k/v live in one packed [S, 3H] buffer; qs = row stride (3H).
// ---------------------------------------------------------------------------
#define KST 72
#define PST 72

__global__ __launch_bounds__(256) void attn_kernel(
    const unsigned short* __restrict__ q, const unsigned short* __restrict__ k,
    const unsigned short* __restrict__ v, int qs, const int* __restrict__ mask,
    unsigned short* __restrict__ o)
{
    __shared__ unsigned short Kt[64 * KST];        // [key][hd]
    __shared__ unsigned short Vt[64 * KST];        // [hd][key]  (transposed)
    __shared__ unsigned short Pt[8][16 * PST];     // per-wave P tile [qrow][key]
    int q0 = blockIdx.x * 128;
    int h  = blockIdx.y;
    int t = threadIdx.x, lane = t & 31, wave = t >> 5;
    int qbase = q0 + wave * 16;
    int fr = lane & 15, hi = lane >> 4, ks = hi * 8;

    // Q fragments for this wave's 16 rows (held in registers for whole kernel)
    FragU qa[2];
    {
        const unsigned short* p = q + (size_t)(qbase + fr) * qs + h * HD;
        #pragma unroll
        for (int kh = 0; kh < 2; ++kh) {
            qa[kh].u[0] = *(const uint4*)(p + kh * 32 + ks);
            qa[kh].u[1] = *(const uint4*)(p + kh * 32 + ks + 16);
        }
    }

    f32x8 zero = {0.f,0.f,0.f,0.f,0.f,0.f,0.f,0.f};
    f32x8 oacc[4];
    float mrow[8], lrow[8];
    #pragma unroll
    for (int d = 0; d < 4; ++d) oacc[d] = zero;
    #pragma unroll
    for (int j = 0; j < 8; ++j) { mrow[j] = -1e30f; lrow[j] = 0.f; }

    const float scale = 0.125f;   // 1/sqrt(64)

    for (int kb = 0; kb < q0 + 128; kb += 64) {
        {   // cooperative load of K tile and transposed V tile
            int r = t >> 2;                 // key 0..63
            int c0 = (t & 3) * 16;          // hd chunk
            const uint4* src = (const uint4*)(k + (size_t)(kb + r) * qs + h * HD + c0);
            uint4* dst = (uint4*)(Kt + r * KST + c0);
            dst[0] = src[0];
            dst[1] = src[1];
            const unsigned short* vs = v + (size_t)(kb + r) * qs + h * HD + c0;
            #pragma unroll
            for (int i = 0; i < 16; ++i)
                Vt[(c0 + i) * KST + r] = vs[i];
        }
        __syncthreads();

        if (kb <= qbase + 15) {
            // S = Q @ K^T  (16 x 64), 4 n-tiles x 2 k-steps
            f32x8 sacc[4];
            #pragma unroll
            for (int nt = 0; nt < 4; ++nt) {
                sacc[nt] = zero;
                #pragma unroll
                for (int kh = 0; kh < 2; ++kh) {
                    FragU bk;
                    const unsigned short* kp = Kt + (nt * 16 + fr) * KST + kh * 32 + ks;
                    bk.u[0] = *(const uint4*)(kp);
                    bk.u[1] = *(const uint4*)(kp + 16);
                    sacc[nt] = __builtin_amdgcn_wmma_f32_16x16x32_bf16(
                        false, qa[kh].v, false, bk.v, (short)0, sacc[nt], false, false);
                }
            }
            // scale + causal + pad mask
            #pragma unroll
            for (int nt = 0; nt < 4; ++nt) {
                int key = kb + nt * 16 + fr;
                int mv = mask[key];
                #pragma unroll
                for (int j = 0; j < 8; ++j) {
                    int mr = qbase + hi * 8 + j;
                    float s = sacc[nt][j] * scale;
                    if (key > mr || mv == 0) s = -1e30f;
                    sacc[nt][j] = s;
                }
            }
            // online softmax (row reductions across 16 lanes of each half)
            #pragma unroll
            for (int j = 0; j < 8; ++j) {
                float rmax = fmaxf(fmaxf(sacc[0][j], sacc[1][j]),
                                   fmaxf(sacc[2][j], sacc[3][j]));
                #pragma unroll
                for (int msk = 8; msk >= 1; msk >>= 1)
                    rmax = fmaxf(rmax, __shfl_xor(rmax, msk));
                float mnew = fmaxf(mrow[j], rmax);
                float alpha = __expf(mrow[j] - mnew);
                mrow[j] = mnew;
                lrow[j] *= alpha;
                #pragma unroll
                for (int d = 0; d < 4; ++d) oacc[d][j] *= alpha;
                float rsum = 0.f;
                #pragma unroll
                for (int nt = 0; nt < 4; ++nt) {
                    float sv = sacc[nt][j];
                    float p = (sv <= -1e29f) ? 0.f : __expf(sv - mnew);
                    sacc[nt][j] = p;
                    rsum += p;
                }
                #pragma unroll
                for (int msk = 8; msk >= 1; msk >>= 1)
                    rsum += __shfl_xor(rsum, msk);
                lrow[j] += rsum;
            }
            // P: C-layout -> LDS -> A-layout fragments
            unsigned short* P = Pt[wave];
            #pragma unroll
            for (int nt = 0; nt < 4; ++nt)
                #pragma unroll
                for (int j = 0; j < 8; ++j)
                    P[(hi * 8 + j) * PST + nt * 16 + fr] = f2bf(sacc[nt][j]);

            FragU pa[2];
            #pragma unroll
            for (int kh = 0; kh < 2; ++kh) {
                const unsigned short* pp = P + fr * PST + kh * 32 + ks;
                pa[kh].u[0] = *(const uint4*)(pp);
                pa[kh].u[1] = *(const uint4*)(pp + 16);
            }
            // O += P @ V  (B frags from transposed V tile)
            #pragma unroll
            for (int dt = 0; dt < 4; ++dt) {
                #pragma unroll
                for (int kh = 0; kh < 2; ++kh) {
                    FragU bv;
                    const unsigned short* vp = Vt + (dt * 16 + fr) * KST + kh * 32 + ks;
                    bv.u[0] = *(const uint4*)(vp);
                    bv.u[1] = *(const uint4*)(vp + 16);
                    oacc[dt] = __builtin_amdgcn_wmma_f32_16x16x32_bf16(
                        false, pa[kh].v, false, bv.v, (short)0, oacc[dt], false, false);
                }
            }
        }
        __syncthreads();
    }

    #pragma unroll
    for (int dt = 0; dt < 4; ++dt)
        #pragma unroll
        for (int j = 0; j < 8; ++j) {
            int row = qbase + hi * 8 + j;
            float val = oacc[dt][j] / lrow[j];
            o[(size_t)row * H + h * HD + dt * 16 + fr] = f2bf(val);
        }
}

// ---------------------------------------------------------------------------
// Host orchestration
// ---------------------------------------------------------------------------
extern "C" void kernel_launch(void* const* d_in, const int* in_sizes, int n_in,
                              void* d_out, int out_size, void* d_ws, size_t ws_size,
                              hipStream_t stream) {
    const float* x   = (const float*)d_in[0];
    const int*   msk = (const int*)d_in[1];
    const float* wq  = (const float*)d_in[2];
    const float* wk  = (const float*)d_in[3];
    const float* wv  = (const float*)d_in[4];
    const float* wo  = (const float*)d_in[5];
    const float* w1  = (const float*)d_in[6];
    const float* b1  = (const float*)d_in[7];
    const float* w2  = (const float*)d_in[8];
    const float* b2  = (const float*)d_in[9];
    const float* g1  = (const float*)d_in[10];
    const float* be1 = (const float*)d_in[11];
    const float* g2  = (const float*)d_in[12];
    const float* be2 = (const float*)d_in[13];

    char* ws = (char*)d_ws;
    size_t off = 0;
    auto take = [&](size_t bytes) { char* p = ws + off; off += (bytes + 255) & ~(size_t)255; return p; };

    const size_t HH2  = (size_t)H * H * 2;
    const size_t HFF2 = (size_t)H * FF * 2;
    const size_t SH2  = (size_t)S * H * 2;
    const size_t SH4  = (size_t)S * H * 4;

    // wqT/wkT/wvT contiguous -> packed QKV weight [3H, H] (bf16, transposed)
    unsigned short* wqkvT = (unsigned short*)take(3 * HH2);
    unsigned short* wqT = wqkvT;
    unsigned short* wkT = wqkvT + (size_t)H * H;
    unsigned short* wvT = wqkvT + 2 * (size_t)H * H;
    unsigned short* woT = (unsigned short*)take(HH2);
    unsigned short* w1T = (unsigned short*)take(HFF2);
    unsigned short* w2T = (unsigned short*)take(HFF2);
    unsigned short* hln = (unsigned short*)take(SH2);          // reused for LN2
    unsigned short* qkv = (unsigned short*)take(3 * SH2);      // packed [S, 3H]
    unsigned short* ao  = (unsigned short*)take(SH2);
    unsigned short* hff = qkv;                                  // S*FF*2 == 4*SH2
    float* x1 = (float*)take(SH4);
    (void)ws_size; (void)in_sizes; (void)n_in; (void)out_size;

    dim3 blk(256);

    // weight convert + transpose to bf16 [N,K]
    wconvT_kernel<<<dim3(H / 32,  H / 32),  blk, 0, stream>>>(wq, wqT, H, H);
    wconvT_kernel<<<dim3(H / 32,  H / 32),  blk, 0, stream>>>(wk, wkT, H, H);
    wconvT_kernel<<<dim3(H / 32,  H / 32),  blk, 0, stream>>>(wv, wvT, H, H);
    wconvT_kernel<<<dim3(H / 32,  H / 32),  blk, 0, stream>>>(wo, woT, H, H);
    wconvT_kernel<<<dim3(FF / 32, H / 32),  blk, 0, stream>>>(w1, w1T, H, FF);
    wconvT_kernel<<<dim3(H / 32,  FF / 32), blk, 0, stream>>>(w2, w2T, FF, H);

    // LN1
    lnorm_kernel<<<S, blk, 0, stream>>>(x, g1, be1, hln);

    // fused QKV projection: [S,H] @ [H,3H] -> packed [S,3H]
    gemm_bf16_kernel<<<dim3(3 * H / 128, S / 128), blk, 0, stream>>>(
        hln, wqkvT, S, 3 * H, H, 0, nullptr, nullptr, qkv, nullptr);

    // causal flash attention (q/k/v packed, row stride 3H)
    attn_kernel<<<dim3(S / 128, NH), blk, 0, stream>>>(
        qkv, qkv + H, qkv + 2 * H, 3 * H, msk, ao);

    // output projection + residual -> x1 (fp32)
    gemm_bf16_kernel<<<dim3(H / 128, S / 128), blk, 0, stream>>>(
        ao, woT, S, H, H, 2, nullptr, x, nullptr, x1);

    // LN2
    lnorm_kernel<<<S, blk, 0, stream>>>(x1, g2, be2, hln);

    // FFN1: GELU(x @ w1 + b1) -> bf16
    gemm_bf16_kernel<<<dim3(FF / 128, S / 128), blk, 0, stream>>>(
        hln, w1T, S, FF, H, 1, b1, nullptr, hff, nullptr);

    // FFN2: + b2 + residual -> d_out (fp32)
    gemm_bf16_kernel<<<dim3(H / 128, S / 128), blk, 0, stream>>>(
        hff, w2T, S, H, FF, 2, b2, x1, nullptr, (float*)d_out);
}
